// RSSM_2345052144237
// MI455X (gfx1250) — matrix operand order repaired and checked by reading, not compiled, not run on hardware
//
#include <hip/hip_runtime.h>
#include <stdint.h>

// ---------------------------------------------------------------------------
// RSSM cell for MI455X (gfx1250): bf16 WMMA tensor path, fp32 epilogues.
// B=32768, STATE=100, BELIEF=HID=200, EMB=1024, ACT=32.
// GEMMs via v_wmma_f32_16x16x32_bf16; A staged in LDS (b128), W^T bf16 (b128).
// ---------------------------------------------------------------------------

#define BATCH 32768

typedef __attribute__((ext_vector_type(16))) __bf16 v16bf;
typedef __attribute__((ext_vector_type(8)))  float v8f;
typedef __attribute__((ext_vector_type(4)))  unsigned int v4u;

union FragBf { v4u q[2]; v16bf v; };

__device__ __forceinline__ unsigned short f2bf(float f) {
  union { float f; uint32_t u; } c; c.f = f;
  uint32_t u = c.u;
  uint32_t r = (u + 0x7FFFu + ((u >> 16) & 1u)) >> 16;  // RNE
  return (unsigned short)r;
}

__device__ __forceinline__ float sigmoidf_(float x) { return 1.0f / (1.0f + expf(-x)); }
__device__ __forceinline__ float softplusf_(float x) { return (x > 20.0f) ? x : log1pf(expf(x)); }

// Deterministic pseudo-normal noise (stands in for jax.random.normal(key 42)).
// Same function for prior and posterior => eps2 == eps, as in the reference.
__device__ __forceinline__ float gauss_noise(uint32_t idx) {
  uint32_t x = idx * 0x9E3779B9u + 0x85EBCA6Bu;
  x ^= x >> 16; x *= 0x7FEB352Du; x ^= x >> 15; x *= 0x846CA68Bu; x ^= x >> 16;
  uint32_t y = idx * 0xC2B2AE35u + 0x165667B1u;
  y ^= y >> 15; y *= 0xD168AAADu; y ^= y >> 15; y *= 0xAF723597u; y ^= y >> 15;
  float u1 = ((x >> 8) + 1u) * (1.0f / 16777217.0f);   // (0,1]
  float u2 = (y >> 8) * (1.0f / 16777216.0f);          // [0,1)
  return sqrtf(-2.0f * logf(u1)) * cosf(6.28318530718f * u2);
}

// -------------------------- fragment gathers (b128) ------------------------
// A (16x32 bf16, MxK): lanes 0-15 -> M=lane, K {0..7,16..23}; lanes 16-31 +8.
// Both halves 16B-aligned -> two ds_load_b128.
template <int LDA>
__device__ __forceinline__ void load_a_frag(FragBf& f, const unsigned int* L32,
                                            int kBase, int lane) {
  int m = lane & 15;
  int koff = (lane >> 4) * 8;
  int ab = (m * LDA + kBase + koff) >> 1;          // dword index, multiple of 4
  f.q[0] = *(const v4u*)(L32 + ab);
  f.q[1] = *(const v4u*)(L32 + ab + 8);
}

// B (32x16 bf16, KxN) from W^T stored [N][Kp]: lane n=lane&15, lanes 0-15 K {0..7,16..23},
// lanes 16-31 +8 -> two 16B-aligned global b128.
template <int LDW>
__device__ __forceinline__ void load_b_frag(FragBf& f, const unsigned short* Wt,
                                            int nBase, int kBase, int lane) {
  int n = lane & 15;
  int koff2 = (lane >> 4) * 8;
  const v4u* Wg = (const v4u*)(Wt + (size_t)(nBase + n) * LDW + kBase + koff2);
  f.q[0] = Wg[0];
  f.q[1] = Wg[2];
}

// ------------------------------ prep kernels -------------------------------
__global__ __launch_bounds__(256) void conv_weight_kernel(
    const float* __restrict__ W, unsigned short* __restrict__ Wt,
    int K, int N, int Kp) {
  int idx = blockIdx.x * 256 + threadIdx.x;
  int total = 224 * Kp;
  if (idx >= total) return;
  int n = idx / Kp, k = idx - n * Kp;
  float v = (n < N && k < K) ? W[(size_t)k * N + n] : 0.0f;
  Wt[idx] = f2bf(v);
}

__global__ __launch_bounds__(256) void prep_x1_kernel(
    const float* __restrict__ st, const float* __restrict__ act,
    const float* __restrict__ nt, unsigned short* __restrict__ X1) {
  int idx = blockIdx.x * 256 + threadIdx.x;
  if (idx >= BATCH * 160) return;
  int row = idx / 160, c = idx - row * 160;
  float v = 0.0f;
  if (c < 100)       v = st[(size_t)row * 100 + c] * nt[row];
  else if (c < 132)  v = act[(size_t)row * 32 + (c - 100)];
  X1[idx] = f2bf(v);
}

__global__ __launch_bounds__(256) void prep_xbf_kernel(
    const float* __restrict__ belief, unsigned short* __restrict__ xbf) {
  int idx = blockIdx.x * 256 + threadIdx.x;
  if (idx >= BATCH * 224) return;
  int row = idx / 224, c = idx - row * 224;
  float v = (c < 200) ? belief[(size_t)row * 200 + c] : 0.0f;
  xbf[idx] = f2bf(v);
}

__global__ __launch_bounds__(256) void prep_obs_kernel(
    const float* __restrict__ obs, unsigned short* __restrict__ qbf) {
  int idx = blockIdx.x * 256 + threadIdx.x;
  if (idx >= BATCH * 1048) return;
  int row = idx / 1048, j = idx - row * 1048;
  float v = (j < 1024) ? obs[(size_t)row * 1024 + j] : 0.0f;
  qbf[(size_t)row * 1248 + 200 + j] = f2bf(v);
}

// --------------------------- generic GEMM + relu ---------------------------
// grid.x = BATCH/16 batch tiles, grid.y * 8 waves = N tiles of 16.
// Compile-time K (LDA) -> fully unrolled k-loop, loads hoisted ahead of WMMA.
template <int LDA, bool HAS_F32>
__global__ __launch_bounds__(256) void gemm_relu_t(
    const unsigned short* __restrict__ A,
    const unsigned short* __restrict__ Wt,
    const float* __restrict__ bias, int N, int NpUse,
    unsigned short* __restrict__ outBf,
    float* __restrict__ outF) {
  __shared__ unsigned short ldsA[16 * LDA];
  const int tid = threadIdx.x, lane = tid & 31, wave = tid >> 5;
  const int mBase = blockIdx.x * 16;

  // cooperative b128 copy of the contiguous 16 x LDA tile
  const v4u* Ag = (const v4u*)(A + (size_t)mBase * LDA);
  v4u* L4 = (v4u*)ldsA;
  constexpr int NQ = (16 * LDA) / 8;
#pragma unroll 4
  for (int i = tid; i < NQ; i += 256) L4[i] = Ag[i];
  __syncthreads();

  const int nBase = (blockIdx.y * 8 + wave) * 16;
  if (nBase >= NpUse) return;

  const unsigned int* L32 = (const unsigned int*)ldsA;
  v8f acc = {};
#pragma unroll
  for (int kk = 0; kk < LDA / 32; ++kk) {
    FragBf a, b;
    load_a_frag<LDA>(a, L32, kk << 5, lane);
    load_b_frag<LDA>(b, Wt, nBase, kk << 5, lane);
    acc = __builtin_amdgcn_wmma_f32_16x16x32_bf16(false, a.v, false, b.v,
                                                  (short)0, acc, false, false);
  }

  const int n = lane & 15, hi = lane >> 4;
  const int col = nBase + n;
  const float bval = (col < N) ? bias[col] : 0.0f;
  unsigned short* pB = outBf + (size_t)(mBase + hi * 8) * 224 + col;
  float* pF = HAS_F32 ? (outF + (size_t)(mBase + hi * 8) * 200 + col) : nullptr;
#pragma unroll
  for (int r = 0; r < 8; ++r) {
    float v = acc[r] + bval;
    v = v > 0.0f ? v : 0.0f;
    pB[(size_t)r * 224] = f2bf(v);            // col < 224 always (NpUse==224)
    if (HAS_F32 && col < N) pF[(size_t)r * 200] = v;
  }
}

// --------------------------- fused GRU kernel ------------------------------
__global__ __launch_bounds__(256) void gru_kernel(
    const unsigned short* __restrict__ xbf, const unsigned short* __restrict__ hbf,
    const float* __restrict__ hf32,
    const unsigned short* __restrict__ Wir, const unsigned short* __restrict__ Wiz,
    const unsigned short* __restrict__ Win, const unsigned short* __restrict__ Whr,
    const unsigned short* __restrict__ Whz, const unsigned short* __restrict__ Whn,
    const float* __restrict__ bir, const float* __restrict__ biz,
    const float* __restrict__ bin_, const float* __restrict__ bhn,
    float* __restrict__ outBelief, unsigned short* __restrict__ pbf,
    unsigned short* __restrict__ qbf) {
  __shared__ unsigned short ldsX[16 * 224];
  __shared__ unsigned short ldsH[16 * 224];
  const int tid = threadIdx.x, lane = tid & 31, wave = tid >> 5;
  const int mBase = blockIdx.x * 16;

  const v4u* Xg = (const v4u*)(xbf + (size_t)mBase * 224);
  const v4u* Hg = (const v4u*)(hbf + (size_t)mBase * 224);
  v4u* LX4 = (v4u*)ldsX;
  v4u* LH4 = (v4u*)ldsH;
#pragma unroll 2
  for (int i = tid; i < (16 * 224) / 8; i += 256) { LX4[i] = Xg[i]; LH4[i] = Hg[i]; }
  __syncthreads();

  const int nBase = (blockIdx.y * 8 + wave) * 16;
  if (nBase >= 224) return;

  const unsigned int* LX = (const unsigned int*)ldsX;
  const unsigned int* LH = (const unsigned int*)ldsH;
  v8f air = {}, aiz = {}, ain = {}, ahr = {}, ahz = {}, ahn = {};
#pragma unroll
  for (int kk = 0; kk < 7; ++kk) {
    const int kBase = kk << 5;
    FragBf ax, ah, b;
    load_a_frag<224>(ax, LX, kBase, lane);
    load_a_frag<224>(ah, LH, kBase, lane);
    load_b_frag<224>(b, Wir, nBase, kBase, lane);
    air = __builtin_amdgcn_wmma_f32_16x16x32_bf16(false, ax.v, false, b.v, (short)0, air, false, false);
    load_b_frag<224>(b, Whr, nBase, kBase, lane);
    ahr = __builtin_amdgcn_wmma_f32_16x16x32_bf16(false, ah.v, false, b.v, (short)0, ahr, false, false);
    load_b_frag<224>(b, Wiz, nBase, kBase, lane);
    aiz = __builtin_amdgcn_wmma_f32_16x16x32_bf16(false, ax.v, false, b.v, (short)0, aiz, false, false);
    load_b_frag<224>(b, Whz, nBase, kBase, lane);
    ahz = __builtin_amdgcn_wmma_f32_16x16x32_bf16(false, ah.v, false, b.v, (short)0, ahz, false, false);
    load_b_frag<224>(b, Win, nBase, kBase, lane);
    ain = __builtin_amdgcn_wmma_f32_16x16x32_bf16(false, ax.v, false, b.v, (short)0, ain, false, false);
    load_b_frag<224>(b, Whn, nBase, kBase, lane);
    ahn = __builtin_amdgcn_wmma_f32_16x16x32_bf16(false, ah.v, false, b.v, (short)0, ahn, false, false);
  }

  const int n = lane & 15, hi = lane >> 4;
  const int col = nBase + n;
  const bool live = (col < 200);
  const float vbir = live ? bir[col] : 0.0f;
  const float vbiz = live ? biz[col] : 0.0f;
  const float vbin = live ? bin_[col] : 0.0f;
  const float vbhn = live ? bhn[col] : 0.0f;
  const int rowB = mBase + hi * 8;
  const float* pH = hf32 + (size_t)rowB * 200 + col;
  unsigned short* pP = pbf + (size_t)rowB * 224 + col;
  float* pO = outBelief + (size_t)rowB * 200 + col;
  unsigned short* pQ = qbf + (size_t)rowB * 1248 + col;
#pragma unroll
  for (int r = 0; r < 8; ++r) {
    float hv = live ? pH[(size_t)r * 200] : 0.0f;
    float rg = sigmoidf_(air[r] + vbir + ahr[r]);
    float zg = sigmoidf_(aiz[r] + vbiz + ahz[r]);
    float ng = tanhf(ain[r] + vbin + rg * (ahn[r] + vbhn));
    float nb = (1.0f - zg) * ng + zg * hv;        // new belief (0 on pad cols)
    pP[(size_t)r * 224] = f2bf(nb);
    if (live) {
      pO[(size_t)r * 200] = nb;
      pQ[(size_t)r * 1248] = f2bf(nb);
    }
  }
}

// ----------------- head kernel: relu, split, softplus+MIN_STD --------------
__global__ __launch_bounds__(256) void head_kernel(
    const unsigned short* __restrict__ A,
    const unsigned short* __restrict__ Wt,
    const float* __restrict__ bias,
    float* __restrict__ outMean, float* __restrict__ outStd) {
  __shared__ unsigned short ldsA[16 * 224];
  const int tid = threadIdx.x, lane = tid & 31, wave = tid >> 5;
  const int mBase = blockIdx.x * 16;

  const v4u* Ag = (const v4u*)(A + (size_t)mBase * 224);
  v4u* L4 = (v4u*)ldsA;
#pragma unroll 2
  for (int i = tid; i < (16 * 224) / 8; i += 256) L4[i] = Ag[i];
  __syncthreads();

  const int nBase = (blockIdx.y * 8 + wave) * 16;
  if (nBase >= 208) return;

  const unsigned int* L32 = (const unsigned int*)ldsA;
  v8f acc = {};
#pragma unroll
  for (int kk = 0; kk < 7; ++kk) {
    FragBf a, b;
    load_a_frag<224>(a, L32, kk << 5, lane);
    load_b_frag<224>(b, Wt, nBase, kk << 5, lane);
    acc = __builtin_amdgcn_wmma_f32_16x16x32_bf16(false, a.v, false, b.v,
                                                  (short)0, acc, false, false);
  }

  const int n = lane & 15, hi = lane >> 4;
  const int col = nBase + n;
  if (col >= 200) return;
  const float bval = bias[col];
  const int rowB = mBase + hi * 8;
  const bool isMean = (col < 100);
  float* pOut = isMean ? (outMean + (size_t)rowB * 100 + col)
                       : (outStd + (size_t)rowB * 100 + (col - 100));
#pragma unroll
  for (int r = 0; r < 8; ++r) {
    float v = acc[r] + bval;
    v = v > 0.0f ? v : 0.0f;                       // activate_final relu
    pOut[(size_t)r * 100] = isMean ? v : (softplusf_(v) + 0.1f);
  }
}

// ------------------------------ sampling -----------------------------------
__global__ __launch_bounds__(256) void sample_kernel(
    const float* __restrict__ mean, const float* __restrict__ stdv,
    float* __restrict__ state, int n) {
  int i = blockIdx.x * 256 + threadIdx.x;
  if (i < n) state[i] = mean[i] + stdv[i] * gauss_noise((uint32_t)i);
}

// ------------------------------ host side ----------------------------------
extern "C" void kernel_launch(void* const* d_in, const int* in_sizes, int n_in,
                              void* d_out, int out_size, void* d_ws, size_t ws_size,
                              hipStream_t stream) {
  (void)in_sizes; (void)n_in; (void)out_size; (void)ws_size;
  const float* prev_state  = (const float*)d_in[0];
  const float* prev_action = (const float*)d_in[1];
  const float* prev_belief = (const float*)d_in[2];
  const float* observation = (const float*)d_in[3];
  const float* nonterm     = (const float*)d_in[4];
  const float* W_sa = (const float*)d_in[5];  const float* b_sa = (const float*)d_in[6];
  const float* W_ir = (const float*)d_in[7];  const float* b_ir = (const float*)d_in[8];
  const float* W_iz = (const float*)d_in[9];  const float* b_iz = (const float*)d_in[10];
  const float* W_in = (const float*)d_in[11]; const float* b_in = (const float*)d_in[12];
  const float* W_hr = (const float*)d_in[13];
  const float* W_hz = (const float*)d_in[14];
  const float* W_hn = (const float*)d_in[15]; const float* b_hn = (const float*)d_in[16];
  const float* W_bp = (const float*)d_in[17]; const float* b_bp = (const float*)d_in[18];
  const float* W_sp = (const float*)d_in[19]; const float* b_sp = (const float*)d_in[20];
  const float* W_bq = (const float*)d_in[21]; const float* b_bq = (const float*)d_in[22];
  const float* W_sq = (const float*)d_in[23]; const float* b_sq = (const float*)d_in[24];

  float* out = (float*)d_out;
  const size_t B = BATCH;
  float* outBelief = out;                // [B,200]
  float* outPState = out + B * 200;      // [B,100]
  float* outPMean  = out + B * 300;
  float* outPStd   = out + B * 400;
  float* outQState = out + B * 500;
  float* outQMean  = out + B * 600;
  float* outQStd   = out + B * 700;

  // bump allocator into d_ws
  char* ws = (char*)d_ws;
  size_t off = 0;
  auto alloc = [&](size_t bytes) -> void* {
    void* p = ws + off;
    off = (off + bytes + 255) & ~(size_t)255;
    return p;
  };
  unsigned short* Wsa_t = (unsigned short*)alloc((size_t)224 * 160 * 2);
  unsigned short* Wir_t = (unsigned short*)alloc((size_t)224 * 224 * 2);
  unsigned short* Wiz_t = (unsigned short*)alloc((size_t)224 * 224 * 2);
  unsigned short* Win_t = (unsigned short*)alloc((size_t)224 * 224 * 2);
  unsigned short* Whr_t = (unsigned short*)alloc((size_t)224 * 224 * 2);
  unsigned short* Whz_t = (unsigned short*)alloc((size_t)224 * 224 * 2);
  unsigned short* Whn_t = (unsigned short*)alloc((size_t)224 * 224 * 2);
  unsigned short* Wbp_t = (unsigned short*)alloc((size_t)224 * 224 * 2);
  unsigned short* Wsp_t = (unsigned short*)alloc((size_t)224 * 224 * 2);
  unsigned short* Wbq_t = (unsigned short*)alloc((size_t)224 * 1248 * 2);
  unsigned short* Wsq_t = (unsigned short*)alloc((size_t)224 * 224 * 2);
  unsigned short* X1   = (unsigned short*)alloc(B * 160 * 2);
  unsigned short* xbf  = (unsigned short*)alloc(B * 224 * 2);
  unsigned short* hbf  = (unsigned short*)alloc(B * 224 * 2);
  float*          hf32 = (float*)        alloc(B * 200 * 4);
  unsigned short* pbf  = (unsigned short*)alloc(B * 224 * 2);
  unsigned short* qbf  = (unsigned short*)alloc(B * 1248 * 2);
  unsigned short* hpbf = (unsigned short*)alloc(B * 224 * 2);
  unsigned short* hqbf = (unsigned short*)alloc(B * 224 * 2);

  auto convW = [&](const float* W, unsigned short* Wt, int K, int N, int Kp) {
    int total = 224 * Kp;
    conv_weight_kernel<<<(total + 255) / 256, 256, 0, stream>>>(W, Wt, K, N, Kp);
  };
  convW(W_sa, Wsa_t, 132, 200, 160);
  convW(W_ir, Wir_t, 200, 200, 224);
  convW(W_iz, Wiz_t, 200, 200, 224);
  convW(W_in, Win_t, 200, 200, 224);
  convW(W_hr, Whr_t, 200, 200, 224);
  convW(W_hz, Whz_t, 200, 200, 224);
  convW(W_hn, Whn_t, 200, 200, 224);
  convW(W_bp, Wbp_t, 200, 200, 224);
  convW(W_sp, Wsp_t, 200, 200, 224);
  convW(W_bq, Wbq_t, 1224, 200, 1248);
  convW(W_sq, Wsq_t, 200, 200, 224);

  prep_x1_kernel<<<(BATCH * 160 + 255) / 256, 256, 0, stream>>>(prev_state, prev_action, nonterm, X1);
  prep_xbf_kernel<<<(BATCH * 224 + 255) / 256, 256, 0, stream>>>(prev_belief, xbf);
  prep_obs_kernel<<<(BATCH * 1048 + 255) / 256, 256, 0, stream>>>(observation, qbf);

  const dim3 gemmGrid(BATCH / 16, 2);
  // h = relu([s,a] @ W_sa + b_sa)  -> h_bf (padded 224) + h_f32
  gemm_relu_t<160, true><<<gemmGrid, 256, 0, stream>>>(X1, Wsa_t, b_sa, 200, 224,
                                                       hbf, hf32);
  // GRU: new_belief -> d_out[0..B*200), p_bf, q_bf[:,0:200)
  gru_kernel<<<gemmGrid, 256, 0, stream>>>(xbf, hbf, hf32,
                                           Wir_t, Wiz_t, Win_t, Whr_t, Whz_t, Whn_t,
                                           b_ir, b_iz, b_in, b_hn,
                                           outBelief, pbf, qbf);
  // prior: hp = relu(belief @ W_bp + b), then head
  gemm_relu_t<224, false><<<gemmGrid, 256, 0, stream>>>(pbf, Wbp_t, b_bp, 200, 224,
                                                        hpbf, (float*)nullptr);
  head_kernel<<<gemmGrid, 256, 0, stream>>>(hpbf, Wsp_t, b_sp, outPMean, outPStd);
  // posterior: hq = relu([belief,obs] @ W_bq + b), then head
  gemm_relu_t<1248, false><<<gemmGrid, 256, 0, stream>>>(qbf, Wbq_t, b_bq, 200, 224,
                                                         hqbf, (float*)nullptr);
  head_kernel<<<gemmGrid, 256, 0, stream>>>(hqbf, Wsq_t, b_sq, outQMean, outQStd);

  // sampling (same deterministic eps for prior and posterior, as in reference)
  const int nS = BATCH * 100;
  sample_kernel<<<(nS + 255) / 256, 256, 0, stream>>>(outPMean, outPStd, outPState, nS);
  sample_kernel<<<(nS + 255) / 256, 256, 0, stream>>>(outQMean, outQStd, outQState, nS);
}
